// MyLSTM_46755013984499
// MI455X (gfx1250) — matrix operand order repaired and checked by reading, not compiled, not run on hardware
//
#include <hip/hip_runtime.h>
#include <hip/hip_bf16.h>

// ---------------------------------------------------------------------------
// LSTM (S=1024, B=64, I=H=512) for MI455X / gfx1250 (wave32, WMMA bf16).
//
// Phase 1 (parallel):  x_proj[s,b,:] = x[s,b,:] @ W_ih^T + (b_ih+b_hh)
//   -> register-blocked WMMA GEMM, M=65536, N=2048, K=512, bf16 in / f32 acc.
// Phase 2 (sequential): persistent kernel, 16 WGs (one per 32 h-columns,
//   128 gate columns interleaved i,f,g,o). W_hh slice lives in LDS (128KB
//   bf16) for all 1024 steps; h is staged per-step to LDS (64KB) through the
//   CDNA5 async global->LDS path when available; cell state c stays in
//   registers; grid sync via monotonic atomic barrier.
// ---------------------------------------------------------------------------

typedef __attribute__((ext_vector_type(16))) __bf16 v16bf;
typedef __attribute__((ext_vector_type(8)))  float  v8f;
typedef __attribute__((ext_vector_type(4)))  int    i4v;

union Frag16 { v16bf v; uint4 q[2]; };

#define S_LEN 1024
#define BATCH 64
#define HDIM  512
#define NCOLS 2048          // 4*H
#define G_WG  16            // workgroups in recurrent phase

#if defined(__gfx1250__) && __has_builtin(__builtin_amdgcn_global_load_async_to_lds_b128)
#define USE_ASYNC_LDS 1
#else
#define USE_ASYNC_LDS 0
#endif

__device__ __forceinline__ void async_copy_b128(const void* gsrc, void* ldst) {
#if USE_ASYNC_LDS
  __builtin_amdgcn_global_load_async_to_lds_b128(
      (__attribute__((address_space(1))) i4v*)gsrc,
      (__attribute__((address_space(3))) i4v*)ldst, 0, 0);
#else
  (void)gsrc; (void)ldst;
#endif
}

__device__ __forceinline__ void async_wait0() {
#if USE_ASYNC_LDS
#if __has_builtin(__builtin_amdgcn_s_wait_asynccnt)
  __builtin_amdgcn_s_wait_asynccnt(0);
#else
  asm volatile("s_wait_asynccnt 0" ::: "memory");
#endif
#endif
}

__device__ __forceinline__ unsigned short f2bf(float f) {
  unsigned u = __float_as_uint(f);
  u += 0x7FFFu + ((u >> 16) & 1u);          // round-to-nearest-even
  return (unsigned short)(u >> 16);
}

__device__ __forceinline__ v8f v8f_zero() {
  v8f z;
#pragma unroll
  for (int i = 0; i < 8; ++i) z[i] = 0.0f;
  return z;
}

// ---- convert x (f32 -> bf16) -----------------------------------------------
__global__ void k_cvt_x(const float* __restrict__ x,
                        unsigned short* __restrict__ xb, int n) {
  int i = blockIdx.x * blockDim.x + threadIdx.x;
  int stride = gridDim.x * blockDim.x;
  for (; i < n; i += stride) xb[i] = f2bf(x[i]);
}

// ---- pack weights [2048][512] f32 -> [16][2048][32] bf16 (B-frag native) ---
__global__ void k_pack_w(const float* __restrict__ w,
                         unsigned short* __restrict__ wp) {
  int i = blockIdx.x * blockDim.x + threadIdx.x;
  int stride = gridDim.x * blockDim.x;
  for (; i < NCOLS * HDIM; i += stride) {
    int n = i >> 9;            // 0..2047
    int k = i & 511;           // 0..511
    wp[((size_t)(k >> 5) * NCOLS + n) * 32 + (k & 31)] = f2bf(w[i]);
  }
}

// ---- combined bias ---------------------------------------------------------
__global__ void k_bias(const float* __restrict__ bi, const float* __restrict__ bh,
                       float* __restrict__ bc) {
  int i = blockIdx.x * blockDim.x + threadIdx.x;
  if (i < NCOLS) bc[i] = bi[i] + bh[i];
}

// ---- barrier init ----------------------------------------------------------
__global__ void k_init(unsigned* __restrict__ bar) {
  if (threadIdx.x == 0 && blockIdx.x == 0) bar[0] = 0u;
}

// ---- Phase 1: x_proj GEMM (M=65536, N=2048, K=512) -------------------------
// Block = 256 threads = 8 waves, block tile 128x128, wave tile 64x32.
__global__ __launch_bounds__(256)
void k_xproj(const unsigned short* __restrict__ xb,   // [65536][512] bf16
             const unsigned short* __restrict__ wih,  // [16][2048][32] bf16
             const float* __restrict__ bias,          // [2048]
             float* __restrict__ xp)                  // [65536][2048] f32
{
  const int lane = threadIdx.x & 31;
  const int w    = threadIdx.x >> 5;      // 0..7
  const int wm   = w >> 2;                // 0..1
  const int wn   = w & 3;                 // 0..3
  const int bm   = blockIdx.x >> 4;       // 0..511
  const int bn   = blockIdx.x & 15;       // 0..15
  const int m0   = bm * 128 + wm * 64;
  const int n0   = bn * 128 + wn * 32;
  const int r    = lane & 15;
  const int half = lane >> 4;

  v8f acc[4][2];
#pragma unroll
  for (int mt = 0; mt < 4; ++mt)
#pragma unroll
    for (int nt = 0; nt < 2; ++nt) acc[mt][nt] = v8f_zero();

  for (int kb = 0; kb < 16; ++kb) {
    Frag16 a[4], b[2];
#pragma unroll
    for (int mt = 0; mt < 4; ++mt) {
      const uint4* p = (const uint4*)(xb + (size_t)(m0 + mt * 16 + r) * HDIM +
                                      kb * 32 + half * 8);
      a[mt].q[0] = p[0];          // K = half*8 .. +7
      a[mt].q[1] = p[2];          // K = 16+half*8 .. +7
    }
#pragma unroll
    for (int nt = 0; nt < 2; ++nt) {
      const uint4* p = (const uint4*)(wih +
          ((size_t)kb * NCOLS + (n0 + nt * 16 + r)) * 32 + half * 8);
      b[nt].q[0] = p[0];
      b[nt].q[1] = p[2];
    }
#pragma unroll
    for (int mt = 0; mt < 4; ++mt)
#pragma unroll
      for (int nt = 0; nt < 2; ++nt)
        acc[mt][nt] = __builtin_amdgcn_wmma_f32_16x16x32_bf16(
            false, a[mt].v, false, b[nt].v, (short)0, acc[mt][nt], false, false);
  }

#pragma unroll
  for (int mt = 0; mt < 4; ++mt) {
#pragma unroll
    for (int nt = 0; nt < 2; ++nt) {
      const int col = n0 + nt * 16 + r;
      const float bs = bias[col];
#pragma unroll
      for (int rr = 0; rr < 8; ++rr) {
        const int row = m0 + mt * 16 + half * 8 + rr;
        xp[(size_t)row * NCOLS + col] = acc[mt][nt][rr] + bs;
      }
    }
  }
}

// ---- Phase 2: persistent recurrent kernel ----------------------------------
// Grid = 16 WGs x 256 threads. WG g owns h-cols [g*32, g*32+32).
// Local column cl in [0,128): gate = cl&3 (interleaved), hcol = cl>>2.
// Wave w owns cl = 16w..16w+15 => 4 h-cols x 4 gates => cell update wave-local.
__global__ __launch_bounds__(256)
void k_lstm(const unsigned short* __restrict__ whh,  // [16][2048][32] bf16
            const float* __restrict__ xp,            // [S*B][2048] f32
            unsigned short* __restrict__ hbuf,       // [2][64][512] bf16
            unsigned* __restrict__ bar,
            float* __restrict__ out)                 // [S*B*H] + [B*H] + [B*H]
{
  extern __shared__ char smem[];
  unsigned short* lds_w = (unsigned short*)smem;              // 16*128*32 bf16
  unsigned short* lds_h = (unsigned short*)(smem + 131072);   // 64*512 bf16

  const int g    = blockIdx.x;
  const int tid  = threadIdx.x;
  const int lane = tid & 31;
  const int w    = tid >> 5;
  const int r    = lane & 15;
  const int half = lane >> 4;
  const int gate = lane & 3;
  const int cl   = w * 16 + r;                 // 0..127
  const int ncol = gate * HDIM + g * 32 + (cl >> 2);   // global gate column
  const int hc   = g * 32 + (cl >> 2);                 // global h column
  const int qbase = lane & 0x1C;               // gate-quartet base lane

  // Load W_hh slice (cols {gate*512 + g*32 + 0..31}) into LDS, frag-native.
  for (int ch = tid; ch < 8192; ch += 256) {     // 8192 x 16B = 128KB
    const int kk8 = ch & 3;
    const int c   = (ch >> 2) & 127;
    const int kb  = ch >> 9;
    const int gcol = (c & 3) * HDIM + g * 32 + (c >> 2);
    const size_t sidx = ((size_t)kb * NCOLS + gcol) * 4 + kk8;
#if USE_ASYNC_LDS
    async_copy_b128(&((const uint4*)whh)[sidx], &((uint4*)lds_w)[ch]);
#else
    ((uint4*)lds_w)[ch] = ((const uint4*)whh)[sidx];
#endif
  }
  async_wait0();

  float c_reg[4][8];
#pragma unroll
  for (int mt = 0; mt < 4; ++mt)
#pragma unroll
    for (int rr = 0; rr < 8; ++rr) c_reg[mt][rr] = 0.0f;

  __syncthreads();

  for (int t = 0; t < S_LEN; ++t) {
    // Stage h_{t-1} into LDS (zeros for t==0); double-buffered in global.
    if (t == 0) {
      for (int i = tid; i < 4096; i += 256)
        ((uint4*)lds_h)[i] = make_uint4(0u, 0u, 0u, 0u);
    } else {
      const uint4* src = (const uint4*)(hbuf + (size_t)((t + 1) & 1) * 32768);
#if USE_ASYNC_LDS
      for (int i = tid; i < 4096; i += 256)
        async_copy_b128(&src[i], &((uint4*)lds_h)[i]);
      async_wait0();
#else
      for (int i = tid; i < 4096; i += 256) ((uint4*)lds_h)[i] = src[i];
#endif
    }
    __syncthreads();

    // gates_acc = h_{t-1} @ W_hh^T (this WG's 128 gate columns, wave: 16)
    v8f acc[4];
#pragma unroll
    for (int mt = 0; mt < 4; ++mt) acc[mt] = v8f_zero();

    for (int kb = 0; kb < 16; ++kb) {
      Frag16 b;
      const uint4* pb =
          (const uint4*)(lds_w + ((size_t)kb * 128 + cl) * 32 + half * 8);
      b.q[0] = pb[0];
      b.q[1] = pb[2];
#pragma unroll
      for (int mt = 0; mt < 4; ++mt) {
        Frag16 a;
        const uint4* pa =
            (const uint4*)(lds_h + (mt * 16 + r) * HDIM + kb * 32 + half * 8);
        a.q[0] = pa[0];
        a.q[1] = pa[2];
        acc[mt] = __builtin_amdgcn_wmma_f32_16x16x32_bf16(
            false, a.v, false, b.v, (short)0, acc[mt], false, false);
      }
    }

    // Cell update (wave-local: gather i,f,g,o across the gate quartet).
    const float* xrow = xp + (size_t)t * BATCH * NCOLS;
#pragma unroll
    for (int mt = 0; mt < 4; ++mt) {
#pragma unroll
      for (int rr = 0; rr < 8; ++rr) {
        const int b = mt * 16 + half * 8 + rr;
        const float pre = acc[mt][rr] + xrow[(size_t)b * NCOLS + ncol];
        const float act = (gate == 2) ? tanhf(pre)
                                      : 1.0f / (1.0f + __expf(-pre));
        const float iv = __shfl(act, qbase + 0, 32);
        const float fv = __shfl(act, qbase + 1, 32);
        const float gv = __shfl(act, qbase + 2, 32);
        const float ov = __shfl(act, qbase + 3, 32);
        const float cn = fv * c_reg[mt][rr] + iv * gv;
        c_reg[mt][rr] = cn;
        const float hv = ov * tanhf(cn);
        if (gate == 0) {
          out[((size_t)t * BATCH + b) * HDIM + hc] = hv;
          hbuf[(size_t)(t & 1) * 32768 + b * HDIM + hc] = f2bf(hv);
          if (t == S_LEN - 1) {
            const size_t tail = (size_t)S_LEN * BATCH * HDIM;
            out[tail + (size_t)b * HDIM + hc] = hv;                    // h_n
            out[tail + (size_t)BATCH * HDIM + (size_t)b * HDIM + hc] = cn; // c_n
          }
        }
      }
    }

    // Prefetch next step's x_proj rows into cache while others finish.
    if (t + 1 < S_LEN) {
      const float* xn = xp + (size_t)(t + 1) * BATCH * NCOLS;
#pragma unroll
      for (int mt = 0; mt < 4; ++mt)
        __builtin_prefetch(xn + (size_t)(mt * 16 + half * 8) * NCOLS + ncol, 0, 1);
    }

    // Grid-wide barrier (monotonic counter, one arrival per WG per step).
    // Every thread releases its own outstanding stores before arrival.
    __threadfence();
    __syncthreads();
    if (tid == 0) {
      __hip_atomic_fetch_add(bar, 1u, __ATOMIC_RELEASE, __HIP_MEMORY_SCOPE_AGENT);
      const unsigned target = (unsigned)(t + 1) * G_WG;
      while (__hip_atomic_load(bar, __ATOMIC_ACQUIRE, __HIP_MEMORY_SCOPE_AGENT) <
             target)
        __builtin_amdgcn_s_sleep(2);
      __threadfence();
    }
    __syncthreads();
  }
}

// ---------------------------------------------------------------------------
extern "C" void kernel_launch(void* const* d_in, const int* in_sizes, int n_in,
                              void* d_out, int out_size, void* d_ws, size_t ws_size,
                              hipStream_t stream) {
  (void)in_sizes; (void)n_in; (void)out_size; (void)ws_size;
  const float* x   = (const float*)d_in[0];   // [1024,64,512]
  const float* wih = (const float*)d_in[1];   // [2048,512]
  const float* whh = (const float*)d_in[2];   // [2048,512]
  const float* bih = (const float*)d_in[3];   // [2048]
  const float* bhh = (const float*)d_in[4];   // [2048]
  float* out = (float*)d_out;

  char* ws = (char*)d_ws;
  size_t off = 0;
  unsigned short* xb   = (unsigned short*)(ws + off); off += (size_t)S_LEN * BATCH * HDIM * 2;  // 64MB
  unsigned short* wihp = (unsigned short*)(ws + off); off += (size_t)NCOLS * HDIM * 2;          // 2MB
  unsigned short* whhp = (unsigned short*)(ws + off); off += (size_t)NCOLS * HDIM * 2;          // 2MB
  float*          bc   = (float*)(ws + off);          off += (size_t)NCOLS * 4;
  off = (off + 255) & ~(size_t)255;
  float*          xpj  = (float*)(ws + off);          off += (size_t)S_LEN * BATCH * NCOLS * 4; // 512MB
  unsigned short* hbuf = (unsigned short*)(ws + off); off += (size_t)2 * BATCH * HDIM * 2;
  unsigned*       bar  = (unsigned*)(ws + off);       off += 256;

  k_cvt_x<<<4096, 256, 0, stream>>>(x, xb, S_LEN * BATCH * HDIM);
  k_pack_w<<<1024, 256, 0, stream>>>(wih, wihp);
  k_pack_w<<<1024, 256, 0, stream>>>(whh, whhp);
  k_bias<<<8, 256, 0, stream>>>(bih, bhh, bc);
  k_init<<<1, 32, 0, stream>>>(bar);
  k_xproj<<<8192, 256, 0, stream>>>(xb, wihp, bc, xpj);

  const int smem_bytes = 131072 + 65536;  // W_hh slice + h stage (within 320KB WGP LDS)
  (void)hipFuncSetAttribute((const void*)k_lstm,
                            hipFuncAttributeMaxDynamicSharedMemorySize, smem_bytes);
  k_lstm<<<G_WG, 256, smem_bytes, stream>>>(whhp, xpj, hbuf, bar, out);
}
